// CausalSelfAttention_26560077758608
// MI455X (gfx1250) — matrix operand order repaired and checked by reading, not compiled, not run on hardware
//
#include <hip/hip_runtime.h>
#include <math.h>

// ---------------------------------------------------------------------------
// CDNA5 (gfx1250) causal self-attention, bf16 WMMA pipeline + TDM staging.
// B=4, T=2048, C=1024, H=16, D=64.
// ---------------------------------------------------------------------------

typedef __attribute__((ext_vector_type(16))) __bf16 v16bf;
typedef __attribute__((ext_vector_type(8)))  __bf16 v8bf;
typedef __attribute__((ext_vector_type(8)))  float  v8f;
typedef unsigned int v4u __attribute__((ext_vector_type(4)));
typedef int          v8i __attribute__((ext_vector_type(8)));
typedef int          v4i __attribute__((ext_vector_type(4)));

#define N_EMBD   1024
#define N_HEAD   16
#define HEAD_DIM 64
#define SEQ_T    2048
#define BATCH    4
#define ROWS     (BATCH * SEQ_T)   // 8192

#if defined(__HIP_DEVICE_COMPILE__) && defined(__gfx1250__) && \
    __has_builtin(__builtin_amdgcn_tensor_load_to_lds) &&      \
    __has_builtin(__builtin_amdgcn_s_wait_tensorcnt)
#define USE_TDM 1
#else
#define USE_TDM 0
#endif

__device__ __forceinline__ __bf16 f2bf(float f) {
  unsigned u = __builtin_bit_cast(unsigned, f);
  u += 0x7FFFu + ((u >> 16) & 1u);          // round-to-nearest-even
  unsigned short h = (unsigned short)(u >> 16);
  return __builtin_bit_cast(__bf16, h);
}

#if USE_TDM
// Issue one 2D TDM tile load: global (row-major, stride0 elements of 2B) -> LDS.
// D# layout per CDNA5 ISA ch.8: group0 = {flags/count, lds_addr, global_addr, type},
// group1 = {data_size, tensor dims (== tile dims; tiles never OOB here), tile dims,
// dim0 stride}. Groups 2/3 zero (2D tensor).
__device__ __forceinline__ void tdm_load_2d(unsigned lds_addr, const void* gptr,
                                            unsigned tile_d0, unsigned tile_d1,
                                            unsigned long long stride0) {
  unsigned long long ga = (unsigned long long)(size_t)gptr;
  v4u g0;
  g0.x = 1u;                                                    // count=1, user D#
  g0.y = lds_addr;                                              // LDS byte address
  g0.z = (unsigned)ga;                                          // global_addr[31:0]
  g0.w = (unsigned)((ga >> 32) & 0x01FFFFFFu) | (2u << 30);     // addr[56:32] | type=2
  v8i g1;
  g1[0] = (int)(1u << 16);                                      // data_size = 2 bytes
  g1[1] = (int)((tile_d0 & 0xFFFFu) << 16);                     // tensor_dim0[15:0]
  g1[2] = (int)((tile_d1 & 0xFFFFu) << 16);                     // tensor_dim1[15:0]
  g1[3] = (int)(tile_d0 << 16);                                 // tile_dim0
  g1[4] = (int)tile_d1;                                         // tile_dim1 (tile_dim2=0)
  g1[5] = (int)(unsigned)(stride0 & 0xFFFFFFFFull);             // dim0 stride lo
  g1[6] = (int)(unsigned)(stride0 >> 32);                       // dim0 stride hi
  g1[7] = 0;
  v4i z4 = {0, 0, 0, 0};
#if __clang_major__ >= 23
  v8i z8 = {0, 0, 0, 0, 0, 0, 0, 0};
  __builtin_amdgcn_tensor_load_to_lds(g0, g1, z4, z4, z8, 0);
#else
  __builtin_amdgcn_tensor_load_to_lds(g0, g1, z4, z4, 0);
#endif
}
#endif

__global__ __launch_bounds__(256) void cvt_f32_bf16(const float* __restrict__ in,
                                                    __bf16* __restrict__ out, int n) {
  int i = blockIdx.x * blockDim.x + threadIdx.x;
  if (i < n) out[i] = f2bf(in[i]);
}

// ---------------------------------------------------------------------------
// GEMM: out[M][N] = A[M][K] * W[N][K]^T   (bf16 in, f32 accumulate)
// One wave -> 64x64 tile (4x4 WMMA frags). Block = 8 waves -> 128x256 tile.
// TDM double-buffers the 128x32 A tile and 256x32 B tile into LDS; the inner
// loop is then pure ds_load_b128 + v_wmma with next tile's DMA in flight.
// out_mode: 1 = bf16, head layout [B][H][T][D]; 2 = f32 row-major + bias.
// ---------------------------------------------------------------------------
__global__ __launch_bounds__(256) void gemm_bf16(const __bf16* __restrict__ A,
                                                 const __bf16* __restrict__ W,
                                                 const float* __restrict__ bias,
                                                 __bf16* __restrict__ outb,
                                                 float* __restrict__ outf,
                                                 int M, int N, int K, int out_mode) {
  const int lane = threadIdx.x & 31;
  const int wave = threadIdx.x >> 5;
  const int wm   = wave >> 2;          // 0..1
  const int wn   = wave & 3;           // 0..3
  const int half = lane >> 4;          // 0/1
  const int l16  = lane & 15;
  const int mBase = blockIdx.y * 128 + wm * 64;
  const int nBase = blockIdx.x * 256 + wn * 64;

  v8f acc[4][4] = {};

#if USE_TDM
  __shared__ __bf16 Ash[2][128][32];   // 16 KiB
  __shared__ __bf16 Bsh[2][256][32];   // 32 KiB
  const unsigned ldsA = (unsigned)(size_t)(&Ash[0][0][0]);
  const unsigned ldsB = (unsigned)(size_t)(&Bsh[0][0][0]);
  const unsigned abuf = 128u * 32u * 2u;   // bytes per A buffer
  const unsigned bbuf = 256u * 32u * 2u;   // bytes per B buffer

  if (wave == 0) {   // prologue DMA for k0 = 0 into buffer 0
    tdm_load_2d(ldsA, A + (size_t)(blockIdx.y * 128) * K, 32u, 128u, (unsigned long long)K);
    tdm_load_2d(ldsB, W + (size_t)(blockIdx.x * 256) * K, 32u, 256u, (unsigned long long)K);
  }

  int buf = 0;
  for (int k0 = 0; k0 < K; k0 += 32) {
    if (wave == 0) __builtin_amdgcn_s_wait_tensorcnt(0);  // current tiles landed
    __syncthreads();                                      // ...visible to all waves,
                                                          // and buf^1 reads retired
    if (wave == 0 && k0 + 32 < K) {                       // prefetch next K-step
      tdm_load_2d(ldsA + (buf ^ 1) * abuf,
                  A + (size_t)(blockIdx.y * 128) * K + k0 + 32, 32u, 128u,
                  (unsigned long long)K);
      tdm_load_2d(ldsB + (buf ^ 1) * bbuf,
                  W + (size_t)(blockIdx.x * 256) * K + k0 + 32, 32u, 256u,
                  (unsigned long long)K);
    }
    v16bf a[4], b[4];
#pragma unroll
    for (int i = 0; i < 4; ++i) {
      const __bf16* ap = &Ash[buf][wm * 64 + i * 16 + l16][half * 16];
      v8bf lo = *(const v8bf*)ap;
      v8bf hi = *(const v8bf*)(ap + 8);
      a[i] = __builtin_shufflevector(lo, hi, 0,1,2,3,4,5,6,7,8,9,10,11,12,13,14,15);
    }
#pragma unroll
    for (int j = 0; j < 4; ++j)
      b[j] = *(const v16bf*)&Bsh[buf][wn * 64 + j * 16 + l16][half * 16];
#pragma unroll
    for (int i = 0; i < 4; ++i)
#pragma unroll
      for (int j = 0; j < 4; ++j)
        acc[i][j] = __builtin_amdgcn_wmma_f32_16x16x32_bf16(
            false, a[i], false, b[j], (short)0, acc[i][j], false, false);
    buf ^= 1;
  }
#else
  for (int k0 = 0; k0 < K; k0 += 32) {
    v16bf a[4], b[4];
#pragma unroll
    for (int i = 0; i < 4; ++i) {
      const __bf16* ap = A + (size_t)(mBase + i * 16 + l16) * K + k0 + half * 16;
      v8bf lo = *(const v8bf*)ap;
      v8bf hi = *(const v8bf*)(ap + 8);
      a[i] = __builtin_shufflevector(lo, hi, 0,1,2,3,4,5,6,7,8,9,10,11,12,13,14,15);
    }
#pragma unroll
    for (int j = 0; j < 4; ++j)
      b[j] = *(const v16bf*)(W + (size_t)(nBase + j * 16 + l16) * K + k0 + half * 16);
#pragma unroll
    for (int i = 0; i < 4; ++i)
#pragma unroll
      for (int j = 0; j < 4; ++j)
        acc[i][j] = __builtin_amdgcn_wmma_f32_16x16x32_bf16(
            false, a[i], false, b[j], (short)0, acc[i][j], false, false);
  }
#endif

#pragma unroll
  for (int i = 0; i < 4; ++i)
#pragma unroll
    for (int j = 0; j < 4; ++j)
#pragma unroll
      for (int r = 0; r < 8; ++r) {
        const int m = mBase + i * 16 + half * 8 + r;   // C-tile: row = vgpr + 8*half
        const int n = nBase + j * 16 + l16;
        const float vv = acc[i][j][r];
        if (out_mode == 2) {
          outf[(size_t)m * N + n] = vv + bias[n];
        } else {
          // head layout: m = b*T + t ; n = h*D + d -> [(b*H+h)*T + t]*D + d
          const int b_ = m >> 11, t = m & (SEQ_T - 1);
          const int h  = n >> 6,  d = n & (HEAD_DIM - 1);
          outb[(((size_t)(b_ * N_HEAD + h) * SEQ_T) + t) * HEAD_DIM + d] = f2bf(vv);
        }
      }
}

// ---------------------------------------------------------------------------
// Flash-style causal attention. Grid: (T/64, B*H). Block = 128 thr = 4 waves,
// each wave owns 16 query rows. KV processed in chunks of 32.
// Q/K/V layout: [BH][T][D] bf16. Output y: [B][T][C] bf16.
// ---------------------------------------------------------------------------
__global__ __launch_bounds__(128) void attn_causal(const __bf16* __restrict__ Q,
                                                   const __bf16* __restrict__ Kk,
                                                   const __bf16* __restrict__ V,
                                                   __bf16* __restrict__ Y) {
  __shared__ __bf16 Vt[HEAD_DIM][32];     // V chunk transposed: [d][kv_rel]
  __shared__ __bf16 Pbuf[4][16][32];      // per-wave P tile (row-major)

  const int lane = threadIdx.x & 31;
  const int wave = threadIdx.x >> 5;      // 0..3
  const int half = lane >> 4;
  const int l16  = lane & 15;
  const int bh   = blockIdx.y;
  const int qBase = blockIdx.x * 64 + wave * 16;

  const __bf16* q = Q  + (size_t)bh * SEQ_T * HEAD_DIM;
  const __bf16* k = Kk + (size_t)bh * SEQ_T * HEAD_DIM;
  const __bf16* v = V  + (size_t)bh * SEQ_T * HEAD_DIM;

  // Q fragments: rows qBase..+15, d split into two K=32 chunks
  v16bf aq[2];
#pragma unroll
  for (int c = 0; c < 2; ++c) {
    const __bf16* qp = q + (size_t)(qBase + l16) * HEAD_DIM + c * 32 + half * 16;
    v8bf lo = *(const v8bf*)qp;
    v8bf hi = *(const v8bf*)(qp + 8);
    aq[c] = __builtin_shufflevector(lo, hi, 0,1,2,3,4,5,6,7,8,9,10,11,12,13,14,15);
  }

  v8f O[4] = {};
  float mrun[8], lrun[8];
#pragma unroll
  for (int r = 0; r < 8; ++r) { mrun[r] = -__builtin_inff(); lrun[r] = 0.f; }

  const int kvEnd = blockIdx.x * 64 + 64;   // exclusive; uniform across block

  for (int kv0 = 0; kv0 < kvEnd; kv0 += 32) {
    __syncthreads();
    {   // stage V chunk transposed into LDS (coalesced global read)
      const int trel = threadIdx.x >> 2;          // 0..31
      const int d0   = (threadIdx.x & 3) * 16;    // 0,16,32,48
      const __bf16* vp = v + (size_t)(kv0 + trel) * HEAD_DIM + d0;
#pragma unroll
      for (int e = 0; e < 16; ++e) Vt[d0 + e][trel] = vp[e];
    }
    __syncthreads();

    // ---- S = (Q K^T) * scale, two 16x16 score tiles ----
    v8f s[2];
#pragma unroll
    for (int jt = 0; jt < 2; ++jt) {
      const int kvB = kv0 + jt * 16;
      v16bf kb[2];
#pragma unroll
      for (int c = 0; c < 2; ++c)
        kb[c] = *(const v16bf*)(k + (size_t)(kvB + l16) * HEAD_DIM + c * 32 + half * 16);
      v8f z = {};
      z = __builtin_amdgcn_wmma_f32_16x16x32_bf16(false, aq[0], false, kb[0], (short)0, z, false, false);
      z = __builtin_amdgcn_wmma_f32_16x16x32_bf16(false, aq[1], false, kb[1], (short)0, z, false, false);
#pragma unroll
      for (int r = 0; r < 8; ++r) {
        const int trow = qBase + half * 8 + r;
        const int kcol = kvB + l16;
        z[r] = (kcol <= trow) ? z[r] * 0.125f : -__builtin_inff();
      }
      s[jt] = z;
    }

    // ---- online softmax update ----
#pragma unroll
    for (int r = 0; r < 8; ++r) {
      float vmax = fmaxf(s[0][r], s[1][r]);
#pragma unroll
      for (int off = 1; off < 16; off <<= 1)
        vmax = fmaxf(vmax, __shfl_xor(vmax, off, 16));
      const float mn = fmaxf(mrun[r], vmax);
      const float al = __expf(mrun[r] - mn);      // 0 on first iteration
      const float p0 = __expf(s[0][r] - mn);
      const float p1 = __expf(s[1][r] - mn);
      s[0][r] = p0; s[1][r] = p1;
      float ps = p0 + p1;
#pragma unroll
      for (int off = 1; off < 16; off <<= 1)
        ps += __shfl_xor(ps, off, 16);
      lrun[r] = lrun[r] * al + ps;
      mrun[r] = mn;
#pragma unroll
      for (int j = 0; j < 4; ++j) O[j][r] *= al;
    }

    // ---- P: C-layout -> A-layout via per-wave LDS tile ----
#pragma unroll
    for (int r = 0; r < 8; ++r) {
      const int row = half * 8 + r;
      Pbuf[wave][row][l16]      = f2bf(s[0][r]);
      Pbuf[wave][row][16 + l16] = f2bf(s[1][r]);
    }
    asm volatile("s_wait_dscnt 0" ::: "memory");   // intra-wave DS RAW
    v16bf ap;
    {
      const __bf16* pp = &Pbuf[wave][l16][half * 16];
      v8bf lo = *(const v8bf*)pp;
      v8bf hi = *(const v8bf*)(pp + 8);
      ap = __builtin_shufflevector(lo, hi, 0,1,2,3,4,5,6,7,8,9,10,11,12,13,14,15);
    }

    // ---- O += P V ----
#pragma unroll
    for (int j = 0; j < 4; ++j) {
      const v16bf vb = *(const v16bf*)&Vt[j * 16 + l16][half * 16];
      O[j] = __builtin_amdgcn_wmma_f32_16x16x32_bf16(false, ap, false, vb, (short)0, O[j], false, false);
    }
  }

  // ---- epilogue: normalize, store y in [B][T][C] ----
  const int b_ = bh >> 4, h = bh & (N_HEAD - 1);
#pragma unroll
  for (int j = 0; j < 4; ++j)
#pragma unroll
    for (int r = 0; r < 8; ++r) {
      const int t = qBase + half * 8 + r;
      const int d = j * 16 + l16;
      const float outv = O[j][r] / lrun[r];
      Y[((size_t)(b_ * SEQ_T + t) * N_EMBD) + h * HEAD_DIM + d] = f2bf(outv);
    }
}

// ---------------------------------------------------------------------------
// Host launcher
// ---------------------------------------------------------------------------
extern "C" void kernel_launch(void* const* d_in, const int* in_sizes, int n_in,
                              void* d_out, int out_size, void* d_ws, size_t ws_size,
                              hipStream_t stream) {
  const float* x  = (const float*)d_in[0];
  const float* Wk = (const float*)d_in[1];
  const float* Wq = (const float*)d_in[2];
  const float* Wv = (const float*)d_in[3];
  const float* Wp = (const float*)d_in[4];
  const float* bp = (const float*)d_in[5];
  float* out = (float*)d_out;

  char* ws = (char*)d_ws;
  const size_t XB   = (size_t)ROWS * N_EMBD * sizeof(__bf16);      // 16 MiB
  const size_t WB   = (size_t)N_EMBD * N_EMBD * sizeof(__bf16);    //  2 MiB
  __bf16* xb  = (__bf16*)(ws);
  __bf16* wkb = (__bf16*)(ws + XB);
  __bf16* wqb = (__bf16*)(ws + XB + WB);
  __bf16* wvb = (__bf16*)(ws + XB + 2 * WB);
  __bf16* wpb = (__bf16*)(ws + XB + 3 * WB);
  __bf16* qb  = (__bf16*)(ws + XB + 4 * WB);
  __bf16* kb  = (__bf16*)(ws + XB + 4 * WB + XB);
  __bf16* vb  = (__bf16*)(ws + XB + 4 * WB + 2 * XB);
  __bf16* yb  = (__bf16*)(ws + XB + 4 * WB + 3 * XB);
  // total workspace: 5*XB + 4*WB = 88 MiB

  const int nX = ROWS * N_EMBD;        // 8388608
  const int nW = N_EMBD * N_EMBD;      // 1048576
  cvt_f32_bf16<<<(nX + 255) / 256, 256, 0, stream>>>(x,  xb,  nX);
  cvt_f32_bf16<<<(nW + 255) / 256, 256, 0, stream>>>(Wk, wkb, nW);
  cvt_f32_bf16<<<(nW + 255) / 256, 256, 0, stream>>>(Wq, wqb, nW);
  cvt_f32_bf16<<<(nW + 255) / 256, 256, 0, stream>>>(Wv, wvb, nW);
  cvt_f32_bf16<<<(nW + 255) / 256, 256, 0, stream>>>(Wp, wpb, nW);

  dim3 ggrid(N_EMBD / 256, ROWS / 128);   // (4, 64)
  gemm_bf16<<<ggrid, 256, 0, stream>>>(xb, wqb, nullptr, qb, nullptr, ROWS, N_EMBD, N_EMBD, 1);
  gemm_bf16<<<ggrid, 256, 0, stream>>>(xb, wkb, nullptr, kb, nullptr, ROWS, N_EMBD, N_EMBD, 1);
  gemm_bf16<<<ggrid, 256, 0, stream>>>(xb, wvb, nullptr, vb, nullptr, ROWS, N_EMBD, N_EMBD, 1);

  attn_causal<<<dim3(SEQ_T / 64, BATCH * N_HEAD), 128, 0, stream>>>(qb, kb, vb, yb);

  gemm_bf16<<<ggrid, 256, 0, stream>>>(yb, wpb, bp, nullptr, out, ROWS, N_EMBD, N_EMBD, 2);
}
